// SimCLRLoss_65146063945894
// MI455X (gfx1250) — matrix operand order repaired and checked by reading, not compiled, not run on hardware
//
#include <hip/hip_runtime.h>
#include <hip/hip_bf16.h>

typedef __attribute__((ext_vector_type(16))) _Float16 v16h;
typedef __attribute__((ext_vector_type(8)))  _Float16 v8h;
typedef __attribute__((ext_vector_type(8)))  float    v8f;
typedef int vint4 __attribute__((vector_size(16)));

#define DIM    256
#define INV_T  5.0f          // 1 / 0.2
#define SHIFT  5.0f          // max possible |sim|/T for unit vectors

#define AS1 __attribute__((address_space(1)))
#define AS3 __attribute__((address_space(3)))

#if defined(__gfx1250__) && __has_builtin(__builtin_amdgcn_global_load_async_to_lds_b128)
#define USE_ASYNC_LDS 1
#else
#define USE_ASYNC_LDS 0
#endif

#if defined(__gfx1250__) && __has_builtin(__builtin_amdgcn_s_wait_asynccnt)
#define WAIT_ASYNC16() __builtin_amdgcn_s_wait_asynccnt(16)
#elif defined(__gfx1250__)
#define WAIT_ASYNC16() asm volatile("s_wait_asynccnt 0x10" ::: "memory")
#else
#define WAIT_ASYNC16()
#endif

#if defined(__gfx1250__) && __has_builtin(__builtin_amdgcn_s_wait_dscnt)
#define WAIT_DS0() __builtin_amdgcn_s_wait_dscnt(0)
#elif defined(__gfx1250__)
#define WAIT_DS0() asm volatile("s_wait_dscnt 0x0" ::: "memory")
#else
#define WAIT_DS0()
#endif

// LDS tile: 16 rows x 512B, each row padded to 544B to spread LDS banks
#define TILE_ROW_B  544
#define TILE_LDS    (16 * TILE_ROW_B)          // 8704 B per tile buffer
#define LSE_DYN_LDS (8 * 2 * TILE_LDS)         // 8 waves x ping-pong

// ---------------------------------------------------------------------------
// Kernel 1: row L2-normalize [Z1;Z2] and convert to f16 (one wave per row)
// ---------------------------------------------------------------------------
__global__ void simclr_nrm_kernel(const float* __restrict__ Z1,
                                  const float* __restrict__ Z2,
                                  _Float16* __restrict__ Zh, int N) {
  const int wave = threadIdx.x >> 5;
  const int lane = threadIdx.x & 31;
  const int row  = blockIdx.x * 8 + wave;                 // 0..2N-1
  const float* src = (row < N) ? (Z1 + (size_t)row * DIM)
                               : (Z2 + (size_t)(row - N) * DIM);
  float v[8];
  float ss = 0.f;
#pragma unroll
  for (int e = 0; e < 8; ++e) { v[e] = src[lane * 8 + e]; ss += v[e] * v[e]; }
#pragma unroll
  for (int off = 16; off; off >>= 1) ss += __shfl_xor(ss, off, 32);
  const float inv = 1.0f / fmaxf(sqrtf(ss), 1e-12f);
  v8h o;
#pragma unroll
  for (int e = 0; e < 8; ++e) o[e] = (_Float16)(v[e] * inv);
  *(v8h*)(Zh + (size_t)row * DIM + lane * 8) = o;
}

// ---------------------------------------------------------------------------
// Kernel 2: fused WMMA similarity + shifted sum-of-exp per row (UNMASKED:
// the identity-diagonal term is subtracted later in kernel 3).
// Block = 256 thr (8 waves) = 32 rows; each wave: 2 A-tiles x (M/8) columns.
// B tiles are staged to a per-wave ping-pong LDS buffer with async global->LDS
// copies (ASYNCcnt): tile ct+1 streams in while tile ct is consumed via
// ds_load_b128 + v_wmma. ssum[i] = sum_j exp(sim_ij/T - SHIFT).
// ---------------------------------------------------------------------------
__global__ void __launch_bounds__(256)
simclr_lse_kernel(const _Float16* __restrict__ Zh,
                  float* __restrict__ ssum, int M) {
  const int wave = threadIdx.x >> 5;
  const int lane = threadIdx.x & 31;
  const int half = lane >> 4;       // 0 or 1
  const int l16  = lane & 15;
  const int row_base = blockIdx.x * 32;

  __shared__ float part_s[8][32];

  // --- A fragments for two 16-row tiles, all K=0..255 ---
  // A 16x32 in contiguous k order (B matches): lanes 0-15 hold K {0..15}, lanes 16-31 hold
  // K {16..31} of row (lane&15).
  v16h a0[8], a1[8];
#pragma unroll
  for (int t = 0; t < 2; ++t) {
    const _Float16* arow =
        Zh + (size_t)(row_base + t * 16 + l16) * DIM + half * 16;
#pragma unroll
    for (int kc = 0; kc < 8; ++kc) {
      v8h lo = *(const v8h*)(arow + kc * 32);
      v8h hi = *(const v8h*)(arow + kc * 32 + 8);
      v16h f = __builtin_shufflevector(lo, hi,
                 0,1,2,3,4,5,6,7,8,9,10,11,12,13,14,15);
      if (t == 0) a0[kc] = f; else a1[kc] = f;
    }
  }

  float s0[8], s1[8];
#pragma unroll
  for (int r = 0; r < 8; ++r) { s0[r] = 0.f; s1[r] = 0.f; }

  const int cols_per_wave = M >> 3;       // 1024
  const int ntiles = cols_per_wave >> 4;  // 64 (power of two)
  const int tmask  = ntiles - 1;
  const int col0   = wave * cols_per_wave;

#if USE_ASYNC_LDS
  extern __shared__ char smem_raw[];
  char* lbase = smem_raw + wave * (2 * TILE_LDS);

  // issue 16 async b128 copies: one 8KB B tile (16 rows x 512B), padded rows
  auto issue_tile = [&](int ct, int buf) {
    const char* g = (const char*)(Zh + (size_t)(col0 + ct * 16) * DIM)
                    + lane * 16;
    char* l = lbase + buf * TILE_LDS + lane * 16;
#pragma unroll
    for (int i = 0; i < 16; ++i)
      __builtin_amdgcn_global_load_async_to_lds_b128(
          (AS1 vint4*)(g + i * 512), (AS3 vint4*)(l + i * TILE_ROW_B),
          0, 0);
  };

  issue_tile(0, 0);
  for (int ct = 0; ct < ntiles; ++ct) {
    // WAR guard: our prior ds_loads of the buffer being overwritten are done
    WAIT_DS0();
    issue_tile((ct + 1) & tmask, (ct + 1) & 1);   // stream next tile
    WAIT_ASYNC16();                               // tile ct has landed
    // B 32x16, same contiguous k order as A: lane = column (lane&15), K {(lane>>4)*16..+15}
    const char* lb = lbase + (ct & 1) * TILE_LDS + l16 * TILE_ROW_B
                     + half * 32;
    v8f c0 = {}, c1 = {};
#pragma unroll
    for (int kc = 0; kc < 8; ++kc) {
      v16h b = *(const v16h*)(lb + kc * 64);      // ds_load_b128 x2
      c0 = __builtin_amdgcn_wmma_f32_16x16x32_f16(
             false, a0[kc], false, b, (short)0, c0, false, false);
      c1 = __builtin_amdgcn_wmma_f32_16x16x32_f16(
             false, a1[kc], false, b, (short)0, c1, false, false);
    }
#pragma unroll
    for (int r = 0; r < 8; ++r) {
      s0[r] += __expf(c0[r] * INV_T - SHIFT);
      s1[r] += __expf(c1[r] * INV_T - SHIFT);
    }
  }
#else
  // Fallback: direct global->VGPR B chunks
  const _Float16* bcol = Zh + (size_t)(col0 + l16) * DIM + half * 16;
  for (int ct = 0; ct < ntiles; ++ct) {
    const _Float16* brow = bcol + (size_t)(ct * 16) * DIM;
    if (ct + 1 < ntiles)
      __builtin_prefetch(brow + 16 * DIM, 0, 1);
    v8f c0 = {}, c1 = {};
#pragma unroll
    for (int kc = 0; kc < 8; ++kc) {
      v16h b = *(const v16h*)(brow + kc * 32);
      c0 = __builtin_amdgcn_wmma_f32_16x16x32_f16(
             false, a0[kc], false, b, (short)0, c0, false, false);
      c1 = __builtin_amdgcn_wmma_f32_16x16x32_f16(
             false, a1[kc], false, b, (short)0, c1, false, false);
    }
#pragma unroll
    for (int r = 0; r < 8; ++r) {
      s0[r] += __expf(c0[r] * INV_T - SHIFT);
      s1[r] += __expf(c1[r] * INV_T - SHIFT);
    }
  }
#endif

  // reduce partial sums across the 16 lanes of each half-wave (fixed order)
#pragma unroll
  for (int r = 0; r < 8; ++r) {
    float x0 = s0[r], x1 = s1[r];
#pragma unroll
    for (int off = 1; off < 16; off <<= 1) {
      x0 += __shfl_xor(x0, off, 32);
      x1 += __shfl_xor(x1, off, 32);
    }
    s0[r] = x0; s1[r] = x1;
  }
  if (l16 == 0) {
#pragma unroll
    for (int r = 0; r < 8; ++r) {
      part_s[wave][half * 8 + r]      = s0[r];   // rows row_base + 0..15
      part_s[wave][16 + half * 8 + r] = s1[r];   // rows row_base + 16..31
    }
  }
  __syncthreads();

  // combine the 8 wave partials per row (deterministic sequential order)
  if (threadIdx.x < 32) {
    float s = 0.f;
#pragma unroll
    for (int w = 0; w < 8; ++w) s += part_s[w][threadIdx.x];
    ssum[row_base + threadIdx.x] = s;
  }
}

// ---------------------------------------------------------------------------
// Kernel 3: selfdot_i = dot(Zn_i,Zn_i), pos_i = dot(Zn_i, Zn_{(i+N)%M})/T
// diff_i = SHIFT + log(ssum_i - exp(selfdot_i/T - SHIFT)) - pos_i
// ---------------------------------------------------------------------------
__global__ void simclr_pos_kernel(const _Float16* __restrict__ Zh,
                                  const float* __restrict__ ssum,
                                  float* __restrict__ diff, int N) {
  const int wave = threadIdx.x >> 5;
  const int lane = threadIdx.x & 31;
  const int M = 2 * N;
  const int i = blockIdx.x * 8 + wave;
  const int j = (i + N) % M;
  v8h xi = *(const v8h*)(Zh + (size_t)i * DIM + lane * 8);
  v8h xj = *(const v8h*)(Zh + (size_t)j * DIM + lane * 8);
  float selfd = 0.f, posd = 0.f;
#pragma unroll
  for (int e = 0; e < 8; ++e) {
    const float xe = (float)xi[e];
    selfd += xe * xe;
    posd  += xe * (float)xj[e];
  }
#pragma unroll
  for (int off = 16; off; off >>= 1) {
    selfd += __shfl_xor(selfd, off, 32);
    posd  += __shfl_xor(posd,  off, 32);
  }
  if (lane == 0) {
    const float s = ssum[i] - __expf(selfd * INV_T - SHIFT);
    diff[i] = SHIFT + __logf(s) - posd * INV_T;
  }
}

// ---------------------------------------------------------------------------
// Kernel 4: deterministic mean over M values (single block)
// ---------------------------------------------------------------------------
__global__ void simclr_mean_kernel(const float* __restrict__ diff,
                                   float* __restrict__ out, int M) {
  __shared__ float red[256];
  float s = 0.f;
  for (int idx = threadIdx.x; idx < M; idx += 256) s += diff[idx];
  red[threadIdx.x] = s;
  __syncthreads();
  for (int off = 128; off; off >>= 1) {
    if ((int)threadIdx.x < off) red[threadIdx.x] += red[threadIdx.x + off];
    __syncthreads();
  }
  if (threadIdx.x == 0) out[0] = red[0] / (float)M;
}

// ---------------------------------------------------------------------------
extern "C" void kernel_launch(void* const* d_in, const int* in_sizes, int n_in,
                              void* d_out, int out_size, void* d_ws, size_t ws_size,
                              hipStream_t stream) {
  const float* Z1 = (const float*)d_in[0];
  const float* Z2 = (const float*)d_in[1];
  const int N = in_sizes[0] / DIM;   // 4096
  const int M = 2 * N;               // 8192

  // workspace layout: Zh f16 [M][DIM] | ssum f32 [M] | diff f32 [M]
  _Float16* Zh  = (_Float16*)d_ws;
  float*    ssm = (float*)((char*)d_ws + (size_t)M * DIM * sizeof(_Float16));
  float*    dif = ssm + M;

  simclr_nrm_kernel <<<M / 8,  256, 0, stream>>>(Z1, Z2, Zh, N);
  simclr_lse_kernel <<<M / 32, 256, LSE_DYN_LDS, stream>>>(Zh, ssm, M);
  simclr_pos_kernel <<<M / 8,  256, 0, stream>>>(Zh, ssm, dif, N);
  simclr_mean_kernel<<<1,      256, 0, stream>>>(dif, (float*)d_out, M);
}
